// SelfAttention_6880537608751
// MI455X (gfx1250) — matrix operand order repaired and checked by reading, not compiled, not run on hardware
//
#include <hip/hip_runtime.h>
#include <hip/hip_bf16.h>
#include <stdint.h>

typedef __attribute__((ext_vector_type(16))) _Float16 v16h;
typedef __attribute__((ext_vector_type(8)))  _Float16 v8h;
typedef __attribute__((ext_vector_type(8)))  float    v8f;

#define TBATCH 8
#define NPOS   1024
#define CH     256
#define HEADS  8
#define DHEAD  32
#define SCALE_F 0.1f
#define BN_EPS  1e-5f

// ---------------------------------------------------------------- utilities

__device__ __forceinline__ float spikef(float x) {
  // round(clamp(x,0,4))/4, RNE like jnp.round
  return rintf(fminf(fmaxf(x, 0.0f), 4.0f)) * 0.25f;
}

__device__ __forceinline__ v8f wmma16x16x32(v16h a, v16h b, v8f c) {
  return __builtin_amdgcn_wmma_f32_16x16x32_f16(false, a, false, b, (short)0, c,
                                                false, false);
}

// A fragment (16x32 f16, row-major source, ld in halves).
__device__ __forceinline__ v16h load_frag_a(const _Float16* p, int ld, int m0,
                                            int k0, int lane) {
  const int m  = m0 + (lane & 15);
  const int hi = lane >> 4;
  const _Float16* r = p + m * ld + k0 + hi * 8;
  v8h lo = *(const v8h*)(r);
  v8h hj = *(const v8h*)(r + 16);
  return __builtin_shufflevector(lo, hj, 0, 1, 2, 3, 4, 5, 6, 7, 8, 9, 10, 11,
                                 12, 13, 14, 15);
}

// B fragment (32x16 f16) from n-major (transposed) storage, ld in halves.
__device__ __forceinline__ v16h load_frag_b(const _Float16* p, int ld, int n0,
                                            int k0, int lane) {
  const int n  = n0 + (lane & 15);
  const int hi = lane >> 4;
  const _Float16* r = p + n * ld + k0 + hi * 8;
  v8h lo = *(const v8h*)(r);
  v8h hj = *(const v8h*)(r + 16);
  return __builtin_shufflevector(lo, hj, 0, 1, 2, 3, 4, 5, 6, 7, 8, 9, 10, 11,
                                 12, 13, 14, 15);
}

// CDNA5 async global->LDS copy (ASYNCcnt-tracked).
__device__ __forceinline__ void async_load_b128(uint32_t lds_off, const void* g) {
  asm volatile("global_load_async_to_lds_b128 %0, %1, off"
               :: "v"(lds_off), "v"((uint64_t)(uintptr_t)g)
               : "memory");
}
__device__ __forceinline__ void wait_asynccnt0() {
  asm volatile("s_wait_asynccnt 0" ::: "memory");
}

// ---------------------------------------------- kernel 1: fold BN into W, b

__global__ __launch_bounds__(256) void fold_bn(
    const float* __restrict__ w, const float* __restrict__ b,
    const float* __restrict__ g, const float* __restrict__ beta,
    const float* __restrict__ mean, const float* __restrict__ var,
    _Float16* __restrict__ WP, float* __restrict__ BP) {
  const int idx = blockIdx.x * 256 + threadIdx.x;   // 0 .. 4*256*256-1
  const int i = idx >> 16;
  const int o = (idx >> 8) & 255;
  const int c = idx & 255;
  const float inv = g[i * 256 + o] * rsqrtf(var[i * 256 + o] + BN_EPS);
  WP[idx] = (_Float16)(w[idx] * inv);
  if (c == 0)
    BP[i * 256 + o] = (b[i * 256 + o] - mean[i * 256 + o]) * inv + beta[i * 256 + o];
}

// ----------------- kernel 2: X = spike(q+qp); Q/K/V = spike(BN(W X)) (WMMA)

__global__ __launch_bounds__(256) void qkv_spike_gemm(
    const float* __restrict__ query, const float* __restrict__ qpos,
    const _Float16* __restrict__ WP, const float* __restrict__ BP,
    _Float16* __restrict__ Qt, _Float16* __restrict__ Kt,
    _Float16* __restrict__ Vt) {
  __shared__ __align__(16) _Float16 Xt[64][CH + 8];   // X^T tile, f16, padded
  __shared__ __align__(16) float stq[32 * 64];        // raw query slice (c,n)
  __shared__ __align__(16) float stp[32 * 64];        // raw query_pos slice

  const int tb   = blockIdx.x >> 4;
  const int n0   = (blockIdx.x & 15) * 64;
  const int tid  = threadIdx.x;
  const int lane = tid & 31;
  const int wv   = tid >> 5;
  const int hi   = lane >> 4;

  const float* qb = query + tb * CH * NPOS;
  const float* pb = qpos  + tb * CH * NPOS;

  // Phase 0: stage (32ch x 64n) f32 slices via async copies, spike->f16,
  // write transposed into Xt (n-major) so it can feed WMMA B fragments.
  const int e  = tid * 8;           // async-copy mapping: 8 floats / thread
  const int cc = e >> 6;            // channel within slice
  const int nn = e & 63;
  const int cn = tid & 63;          // convert mapping: fixed n, 8 channels
  const int cc0 = (tid >> 6) * 8;
#pragma unroll 1
  for (int ks = 0; ks < 8; ++ks) {
    if (ks) __syncthreads();        // staging buffers free to overwrite
    const float* gq = qb + (ks * 32 + cc) * NPOS + n0 + nn;
    const float* gp = pb + (ks * 32 + cc) * NPOS + n0 + nn;
    async_load_b128((uint32_t)(uintptr_t)&stq[e],     gq);
    async_load_b128((uint32_t)(uintptr_t)&stq[e + 4], gq + 4);
    async_load_b128((uint32_t)(uintptr_t)&stp[e],     gp);
    async_load_b128((uint32_t)(uintptr_t)&stp[e + 4], gp + 4);
    wait_asynccnt0();
    __syncthreads();                // everyone's async data landed in LDS
    v8h pk;
#pragma unroll
    for (int j = 0; j < 8; ++j)
      pk[j] = (_Float16)spikef(stq[(cc0 + j) * 64 + cn] + stp[(cc0 + j) * 64 + cn]);
    *(v8h*)(&Xt[cn][ks * 32 + cc0]) = pk;   // one 16B ds_store per thread
  }
  __syncthreads();

  // Phase 1: per branch, wave wv computes output rows [wv*32, wv*32+32).
  const int m0 = wv * 32;
#pragma unroll 1
  for (int br = 0; br < 3; ++br) {
    const _Float16* Wb = WP + br * CH * CH;
    __builtin_prefetch(Wb + m0 * CH, 0, 2);          // pull W' stream into L2
    v8f acc[2][4] = {};
#pragma unroll 1
    for (int ks = 0; ks < 8; ++ks) {
      v16h a0 = load_frag_a(Wb, CH, m0,      ks * 32, lane);
      v16h a1 = load_frag_a(Wb, CH, m0 + 16, ks * 32, lane);
#pragma unroll
      for (int nt = 0; nt < 4; ++nt) {
        v16h bf = load_frag_b(&Xt[0][0], CH + 8, nt * 16, ks * 32, lane);
        acc[0][nt] = wmma16x16x32(a0, bf, acc[0][nt]);
        acc[1][nt] = wmma16x16x32(a1, bf, acc[1][nt]);
      }
    }
    const float* bb = BP + br * CH;
    _Float16* Ob = (br == 0 ? Qt : (br == 1 ? Kt : Vt)) + tb * NPOS * CH;
#pragma unroll
    for (int mt = 0; mt < 2; ++mt) {
      const int rowbase = m0 + mt * 16 + hi * 8;
      float bias[8];
#pragma unroll
      for (int r = 0; r < 8; ++r) bias[r] = bb[rowbase + r];
#pragma unroll
      for (int nt = 0; nt < 4; ++nt) {
        const int n = n0 + nt * 16 + (lane & 15);
        v8h pk;
#pragma unroll
        for (int r = 0; r < 8; ++r)
          pk[r] = (_Float16)spikef(acc[mt][nt][r] + bias[r]);
        *(v8h*)(Ob + n * CH + rowbase) = pk;   // (n, c) f16, 16B store
      }
    }
  }
}

// --------------------------- kernel 3: S[tb,h] = K_h @ V_h^T  (32x32, f32)

__global__ __launch_bounds__(256) void kv_outer(
    const _Float16* __restrict__ Kt, const _Float16* __restrict__ Vt,
    float* __restrict__ S) {
  const int tb = blockIdx.x >> 3;
  const int h  = blockIdx.x & 7;
  const _Float16* Kb = Kt + tb * NPOS * CH + h * DHEAD;
  const _Float16* Vb = Vt + tb * NPOS * CH + h * DHEAD;
  const int d   = threadIdx.x & 31;
  const int dp0 = threadIdx.x >> 5;    // 0..7; covers d' = dp0 + 8*i
  float a0 = 0.f, a1 = 0.f, a2 = 0.f, a3 = 0.f;
#pragma unroll 4
  for (int n = 0; n < NPOS; ++n) {
    const float v = (float)Vb[n * CH + d];          // coalesced over lanes
    a0 += (float)Kb[n * CH + dp0     ] * v;          // wave-uniform broadcast
    a1 += (float)Kb[n * CH + dp0 +  8] * v;
    a2 += (float)Kb[n * CH + dp0 + 16] * v;
    a3 += (float)Kb[n * CH + dp0 + 24] * v;
  }
  float* Sb = S + blockIdx.x * (DHEAD * DHEAD);      // [d'][d]
  Sb[(dp0     ) * 32 + d] = a0;
  Sb[(dp0 +  8) * 32 + d] = a1;
  Sb[(dp0 + 16) * 32 + d] = a2;
  Sb[(dp0 + 24) * 32 + d] = a3;
}

// -------- kernel 4: A = spike(SCALE*S^T Q); Out = BN3(W3 A) + tmp  (WMMA)

__global__ __launch_bounds__(256) void attn_proj(
    const float* __restrict__ query, const float* __restrict__ qpos,
    const _Float16* __restrict__ WP, const float* __restrict__ BP,
    const _Float16* __restrict__ Qt, const float* __restrict__ S,
    float* __restrict__ out) {
  __shared__ __align__(16) _Float16 St[HEADS][DHEAD][DHEAD + 8]; // SCALE*S^T
  __shared__ __align__(16) _Float16 At[32][CH + 8];              // spike(A)^T

  const int tb   = blockIdx.x >> 5;
  const int n0   = (blockIdx.x & 31) * 32;
  const int tid  = threadIdx.x;
  const int lane = tid & 31;
  const int wv   = tid >> 5;
  const int hi   = lane >> 4;

  // load SCALE * S^T into LDS as f16
  const float* Sg = S + tb * HEADS * DHEAD * DHEAD;
#pragma unroll 4
  for (int j = 0; j < 32; ++j) {
    const int p  = tid * 32 + j;          // 0..8191
    const int h  = p >> 10;
    const int dp = (p >> 5) & 31;
    const int d  = p & 31;
    St[h][d][dp] = (_Float16)(SCALE_F * Sg[p]);
  }
  __syncthreads();

  // GEMM1: wave wv owns head h=wv; Aatt_h(32 x 32) = (SCALE*S_h^T) @ Q_h
  {
    const int h = wv;
    const _Float16* Qb = Qt + tb * NPOS * CH + h * DHEAD;  // n-major, k=d'
    v8f acc[2][2] = {};
    v16h a0 = load_frag_a(&St[h][0][0], DHEAD + 8, 0,  0, lane);
    v16h a1 = load_frag_a(&St[h][0][0], DHEAD + 8, 16, 0, lane);
#pragma unroll
    for (int nt = 0; nt < 2; ++nt) {
      v16h bf = load_frag_b(Qb, CH, n0 + nt * 16, 0, lane);
      acc[0][nt] = wmma16x16x32(a0, bf, acc[0][nt]);
      acc[1][nt] = wmma16x16x32(a1, bf, acc[1][nt]);
    }
#pragma unroll
    for (int mt = 0; mt < 2; ++mt) {
      const int rowbase = h * DHEAD + mt * 16 + hi * 8;
#pragma unroll
      for (int nt = 0; nt < 2; ++nt) {
        const int nl = nt * 16 + (lane & 15);
        v8h pk;
#pragma unroll
        for (int r = 0; r < 8; ++r)
          pk[r] = (_Float16)spikef(acc[mt][nt][r]);
        *(v8h*)(&At[nl][rowbase]) = pk;
      }
    }
  }
  __syncthreads();

  // GEMM2: Out = W3' @ spike(Aatt) + b3' + (query + query_pos)
  {
    const _Float16* W3 = WP + 3 * CH * CH;
    const float*    b3 = BP + 3 * CH;
    const int m0 = wv * 32;
    __builtin_prefetch(W3 + m0 * CH, 0, 2);
    v8f acc[2][2] = {};
#pragma unroll 1
    for (int ks = 0; ks < 8; ++ks) {
      v16h a0 = load_frag_a(W3, CH, m0,      ks * 32, lane);
      v16h a1 = load_frag_a(W3, CH, m0 + 16, ks * 32, lane);
#pragma unroll
      for (int nt = 0; nt < 2; ++nt) {
        v16h bf = load_frag_b(&At[0][0], CH + 8, nt * 16, ks * 32, lane);
        acc[0][nt] = wmma16x16x32(a0, bf, acc[0][nt]);
        acc[1][nt] = wmma16x16x32(a1, bf, acc[1][nt]);
      }
    }
    const float* qb = query + tb * CH * NPOS;
    const float* pb = qpos  + tb * CH * NPOS;
    float*       ob = out   + tb * CH * NPOS;
#pragma unroll
    for (int mt = 0; mt < 2; ++mt) {
      const int rowbase = m0 + mt * 16 + hi * 8;
#pragma unroll
      for (int nt = 0; nt < 2; ++nt) {
        const int n = n0 + nt * 16 + (lane & 15);
#pragma unroll
        for (int r = 0; r < 8; ++r) {
          const int c   = rowbase + r;
          const int idx = c * NPOS + n;      // coalesced across the 16 lanes
          ob[idx] = acc[mt][nt][r] + b3[c] + qb[idx] + pb[idx];
        }
      }
    }
  }
}

// ------------------------------------------------------------------ launch

extern "C" void kernel_launch(void* const* d_in, const int* in_sizes, int n_in,
                              void* d_out, int out_size, void* d_ws,
                              size_t ws_size, hipStream_t stream) {
  (void)in_sizes; (void)n_in; (void)out_size; (void)ws_size;
  const float* query = (const float*)d_in[0];
  // d_in[1] ('key') and d_in[2] ('value') are unused by the reference.
  const float* qpos  = (const float*)d_in[3];
  const float* cw    = (const float*)d_in[4];
  const float* cb    = (const float*)d_in[5];
  const float* g     = (const float*)d_in[6];
  const float* bet   = (const float*)d_in[7];
  const float* mu    = (const float*)d_in[8];
  const float* var   = (const float*)d_in[9];
  float* out = (float*)d_out;

  // workspace layout (total ~12.8 MB)
  char* ws = (char*)d_ws;
  _Float16* WP = (_Float16*)(ws);                              // 512 KiB f16
  float*    BP = (float*)(ws + 524288);                        //   4 KiB f32
  _Float16* Qt = (_Float16*)(ws + 528384);                     //   4 MiB f16
  _Float16* Kt = (_Float16*)(ws + 528384 + 4194304);           //   4 MiB f16
  _Float16* Vt = (_Float16*)(ws + 528384 + 2 * 4194304);       //   4 MiB f16
  float*    Sb = (float*)(ws + 528384 + 3 * 4194304);          // 256 KiB f32

  fold_bn<<<dim3(1024), dim3(256), 0, stream>>>(cw, cb, g, bet, mu, var, WP, BP);
  qkv_spike_gemm<<<dim3(128), dim3(256), 0, stream>>>(query, qpos, WP, BP,
                                                      Qt, Kt, Vt);
  kv_outer<<<dim3(64), dim3(256), 0, stream>>>(Kt, Vt, Sb);
  attn_proj<<<dim3(256), dim3(256), 0, stream>>>(query, qpos, WP, BP, Qt, Sb,
                                                 out);
}
